// SlidingWindowAttention_24292335027123
// MI455X (gfx1250) — matrix operand order repaired and checked by reading, not compiled, not run on hardware
//
#include <hip/hip_runtime.h>

// ---- CDNA5 (gfx1250) wave32 WMMA / TDM types ----
typedef __bf16 bf16;
typedef __attribute__((ext_vector_type(16))) __bf16 v16bf;
typedef __attribute__((ext_vector_type(8)))  __bf16 v8bf;
typedef __attribute__((ext_vector_type(8)))  float  v8f;
typedef __attribute__((ext_vector_type(4)))  unsigned int v4u;
typedef __attribute__((ext_vector_type(8)))  int v8i;
typedef __attribute__((ext_vector_type(4)))  int v4i;

#define EMB     1024
#define HEADS   16
#define HDIM    64
#define SEQ     2048
#define BATCH   2
#define HALFWIN 128
#define KSTEP   32
#define LDSROW  40   // 32 payload elems + 8 pad elems (4 dwords) per row -> conflict-free ds_load_b128

static __device__ __forceinline__ v16bf cat8(v8bf lo, v8bf hi) {
  union { v16bf v; v8bf h[2]; } u;
  u.h[0] = lo; u.h[1] = hi;
  return u.v;
}

// Tensor Data Mover: DMA a rows x cols (bf16) 2-D tile from global into LDS.
// D# per CDNA5 ISA; LDS rows padded 16dw payload + 4dw pad (stride = LDSROW elems).
// This toolchain exposes the 6-arg builtin: (g0 v4u, g1 v8i, g2 v4i, g3 v4i, v8i, cpol).
static __device__ __forceinline__ void tdm_load_2d(unsigned lds_off, const void* gptr,
                                                   unsigned stride_elems,
                                                   unsigned rows, unsigned cols) {
  const unsigned long long ga = (unsigned long long)(size_t)gptr;
  v4u g0;
  g0[0] = 1u;                                                  // count=1 (valid), user mode
  g0[1] = lds_off;                                             // lds_addr (bytes)
  g0[2] = (unsigned)ga;                                        // global_addr[31:0]
  g0[3] = (unsigned)((ga >> 32) & 0x01ffffffu) | (2u << 30);   // global_addr[56:32] | type=2
  v8i g1;
  g1[0] = (int)((1u << 16)        // data_size = 1 -> 2 bytes
              | (1u << 20)        // pad_enable
              | (3u << 22)        // pad_interval code 3 = 16 dwords (one 32-elem row)
              | (3u << 25));      // pad_amount   code 3 = 4 dwords (8 elems)
  g1[1] = (int)((cols & 0xffffu) << 16);                                   // tensor_dim0[15:0]
  g1[2] = (int)(((cols >> 16) & 0xffffu) | ((rows & 0xffffu) << 16));      // dim0 hi | dim1 lo
  g1[3] = (int)(((rows >> 16) & 0xffffu) | ((cols & 0xffffu) << 16));      // dim1 hi | tile_dim0
  g1[4] = (int)(rows & 0xffffu);                                           // tile_dim1 | tile_dim2=0
  g1[5] = (int)stride_elems;                                               // tensor_dim0_stride lo
  g1[6] = 0;                                                               // stride hi | dim1_stride lo
  g1[7] = 0;                                                               // dim1_stride hi
  v4i z4 = {};
  v8i z8 = {};
  __builtin_amdgcn_tensor_load_to_lds(g0, g1, z4, z4, z8, 0);
}

// ---------------- fp32 -> bf16 conversion (4-wide) ----------------
__global__ void __launch_bounds__(256)
cvt_f32_bf16_kernel(const float* __restrict__ src, bf16* __restrict__ dst, int n4) {
  int i = blockIdx.x * blockDim.x + threadIdx.x;
  if (i < n4) {
    float4 f = ((const float4*)src)[i];
    dst[i * 4 + 0] = (bf16)f.x;
    dst[i * 4 + 1] = (bf16)f.y;
    dst[i * 4 + 2] = (bf16)f.z;
    dst[i * 4 + 3] = (bf16)f.w;
  }
}

// ---------------- TDM-staged WMMA GEMM: out = a @ W^T + b ----------------
// 128 threads = 4 waves. Block computes 64(M) x 128(N); wave w owns rows [w*16, w*16+16).
// Per K-step (32): TDM DMAs x panel (64x32) + W panel (128x32) into padded LDS (double buffered),
// wave 0 issues/waits TENSORcnt, workgroup barrier publishes, all waves ds_load operands + 8 WMMAs.
// mode: 0=Q (x0.125, [b,h,s,d])  1=K ([b,h,s,d])  2=V (transposed [b,h,d,s])  3=fp32 out [row,col]
__global__ void __launch_bounds__(128)
gemm_tdm_kernel(const bf16* __restrict__ ab, const bf16* __restrict__ wb,
                const float* __restrict__ bias, bf16* __restrict__ dstb,
                float* __restrict__ dstf, int mode) {
  __shared__ __align__(128) bf16 xs[2][64 * LDSROW];
  __shared__ __align__(128) bf16 wsm[2][128 * LDSROW];

  const int tid  = threadIdx.x;
  const int wv   = tid >> 5;
  const int lane = tid & 31;
  const int half = lane >> 4;
  const int l16  = lane & 15;
  const int R0 = blockIdx.y * 64;    // global row base (b*SEQ + s)
  const int N0 = blockIdx.x * 128;   // output feature base
  const int m0 = wv * 16;            // this wave's row slice inside the block

  v8f zero8 = {};
  v8f acc[8];
#pragma unroll
  for (int t = 0; t < 8; ++t) acc[t] = zero8;

  const int nsteps = EMB / KSTEP;
  if (wv == 0) {   // TDM ignores EXEC; issue once per block from wave 0 (per-wave TENSORcnt)
    tdm_load_2d((unsigned)(size_t)&xs[0][0],  ab + (size_t)R0 * EMB, EMB, 64,  KSTEP);
    tdm_load_2d((unsigned)(size_t)&wsm[0][0], wb + (size_t)N0 * EMB, EMB, 128, KSTEP);
  }

  for (int step = 0; step < nsteps; ++step) {
    const int cur = step & 1;
    if (wv == 0) {
      if (step + 1 < nsteps) {   // prefetch next panels into the other buffer, then wait current
        const int kk = (step + 1) * KSTEP;
        tdm_load_2d((unsigned)(size_t)&xs[cur ^ 1][0],  ab + (size_t)R0 * EMB + kk, EMB, 64,  KSTEP);
        tdm_load_2d((unsigned)(size_t)&wsm[cur ^ 1][0], wb + (size_t)N0 * EMB + kk, EMB, 128, KSTEP);
        __builtin_amdgcn_s_wait_tensorcnt(2);   // TENSORcnt in-order: current pair complete
      } else {
        __builtin_amdgcn_s_wait_tensorcnt(0);
      }
    }
    __syncthreads();

    const bf16* xrow = &xs[cur][(m0 + l16) * LDSROW];
    v16bf A = cat8(*(const v8bf*)(xrow + 8 * half),
                   *(const v8bf*)(xrow + 16 + 8 * half));
#pragma unroll
    for (int t = 0; t < 8; ++t) {
      const bf16* wrow = &wsm[cur][(t * 16 + l16) * LDSROW + 8 * half];
      v16bf B = cat8(*(const v8bf*)wrow, *(const v8bf*)(wrow + 16));
      acc[t] = __builtin_amdgcn_wmma_f32_16x16x32_bf16(false, A, false, B,
                                                       (short)0, acc[t], false, false);
    }
    __syncthreads();   // all waves done reading `cur` before it is refilled next step
  }

  const float scale = (mode == 0) ? 0.125f : 1.0f;  // fold 1/sqrt(HDIM) into Q
#pragma unroll
  for (int t = 0; t < 8; ++t) {
    const int col = N0 + t * 16 + l16;
    const float bv = bias[col];
    const int h = col >> 6, d = col & (HDIM - 1);
#pragma unroll
    for (int r = 0; r < 8; ++r) {
      const int row = R0 + m0 + r + 8 * half;
      const float v = (acc[t][r] + bv) * scale;
      if (mode == 3) {
        dstf[(size_t)row * EMB + col] = v;
      } else {
        const int bi = row >> 11;           // row / SEQ
        const int s  = row & (SEQ - 1);
        size_t idx;
        if (mode == 2) idx = (((size_t)(bi * HEADS + h)) * HDIM + d) * SEQ + s;  // V^T
        else           idx = (((size_t)(bi * HEADS + h)) * SEQ + s) * HDIM + d;  // Q,K
        dstb[idx] = (bf16)v;
      }
    }
  }
}

// ---------------- sliding-window attention ----------------
// One wave per (batch, head, 16-query tile). Online softmax over 32-key chunks.
// S^T = K(A) x Q^T(B): C-layout lane n = query n  -> softmax reduce = in-lane + shfl_xor(16)
// P^T C-layout == P A-layout element-for-element  -> free repack to bf16 A operand
// O = P(A) x V(B), V read from pre-transposed [d,s] storage (contiguous per lane)
__global__ void __launch_bounds__(32)
swa_attn_kernel(const bf16* __restrict__ Qb, const bf16* __restrict__ Kb,
                const bf16* __restrict__ Vtb, bf16* __restrict__ attnb) {
  const int lane = threadIdx.x & 31;
  const int half = lane >> 4;
  const int l16  = lane & 15;
  const int q0 = blockIdx.x * 16;
  const int h  = blockIdx.y;
  const int b  = blockIdx.z;

  const bf16* Qh = Qb  + ((size_t)(b * HEADS + h)) * SEQ * HDIM;
  const bf16* Kh = Kb  + ((size_t)(b * HEADS + h)) * SEQ * HDIM;
  const bf16* Vh = Vtb + ((size_t)(b * HEADS + h)) * HDIM * SEQ;

  v16bf QB[2];
#pragma unroll
  for (int c = 0; c < 2; ++c) {
    const bf16* p = Qh + (size_t)(q0 + l16) * HDIM + c * 32 + 16 * half;
    QB[c] = cat8(*(const v8bf*)p, *(const v8bf*)(p + 8));
  }

  v8f zero8 = {};
  v8f o[4];
#pragma unroll
  for (int t = 0; t < 4; ++t) o[t] = zero8;
  float m = -1e30f, lsum = 0.0f;

  int kc_start = q0 - HALFWIN;
  if (kc_start < 0) kc_start = 0;
  kc_start &= ~31;
  int kc_lim = q0 + 16 + HALFWIN;
  if (kc_lim > SEQ) kc_lim = SEQ;

  const int query = q0 + l16;

  for (int kc = kc_start; kc < kc_lim; kc += 32) {
    v8f st[2];
    st[0] = zero8; st[1] = zero8;
#pragma unroll
    for (int sub = 0; sub < 2; ++sub) {
      int krow = kc + sub * 16 + l16;
      if (krow > SEQ - 1) krow = SEQ - 1;
      const bf16* kp = Kh + (size_t)krow * HDIM;
#pragma unroll
      for (int c = 0; c < 2; ++c) {
        const bf16* p = kp + c * 32 + 16 * half;
        v16bf KA = cat8(*(const v8bf*)p, *(const v8bf*)(p + 8));
        st[sub] = __builtin_amdgcn_wmma_f32_16x16x32_bf16(false, KA, false, QB[c],
                                                          (short)0, st[sub], false, false);
      }
    }

    float pv[16];
    float tmax = -1e30f;
#pragma unroll
    for (int sub = 0; sub < 2; ++sub) {
#pragma unroll
      for (int r = 0; r < 8; ++r) {
        const int key = kc + sub * 16 + r + 8 * half;
        float s = st[sub][r];
        const int dlt = query - key;
        if (dlt < -HALFWIN || dlt > HALFWIN) s = -1e30f;
        pv[sub * 8 + r] = s;
        tmax = fmaxf(tmax, s);
      }
    }
    tmax = fmaxf(tmax, __shfl_xor(tmax, 16, 32));
    const float mnew  = fmaxf(m, tmax);
    const float alpha = __expf(m - mnew);

    float psum = 0.0f;
    v16bf PA;
#pragma unroll
    for (int i = 0; i < 16; ++i) {
      const float p = __expf(pv[i] - mnew);
      psum += p;
      PA[i] = (bf16)p;
    }
    psum += __shfl_xor(psum, 16, 32);
    lsum = lsum * alpha + psum;
    m = mnew;

    float af[8];
#pragma unroll
    for (int r = 0; r < 8; ++r) af[r] = __shfl(alpha, r + 8 * half, 32);

#pragma unroll
    for (int t = 0; t < 4; ++t) {
      const bf16* vp = Vh + (size_t)(t * 16 + l16) * SEQ + kc + 8 * half;
      v16bf VB = cat8(*(const v8bf*)vp, *(const v8bf*)(vp + 16));
#pragma unroll
      for (int r = 0; r < 8; ++r) o[t][r] *= af[r];
      o[t] = __builtin_amdgcn_wmma_f32_16x16x32_bf16(false, PA, false, VB,
                                                     (short)0, o[t], false, false);
    }
  }

  float lf[8];
#pragma unroll
  for (int r = 0; r < 8; ++r) lf[r] = __shfl(lsum, r + 8 * half, 32);
#pragma unroll
  for (int t = 0; t < 4; ++t) {
#pragma unroll
    for (int r = 0; r < 8; ++r) {
      const int row = q0 + r + 8 * half;
      const int col = h * HDIM + t * 16 + l16;
      attnb[((size_t)(b * SEQ + row)) * EMB + col] = (bf16)(o[t][r] / lf[r]);
    }
  }
}

// ---------------- host launcher ----------------
extern "C" void kernel_launch(void* const* d_in, const int* in_sizes, int n_in,
                              void* d_out, int out_size, void* d_ws, size_t ws_size,
                              hipStream_t stream) {
  const float* x  = (const float*)d_in[0];
  const float* Wq = (const float*)d_in[1];
  const float* bq = (const float*)d_in[2];
  const float* Wk = (const float*)d_in[3];
  const float* bk = (const float*)d_in[4];
  const float* Wv = (const float*)d_in[5];
  const float* bv = (const float*)d_in[6];
  const float* Wo = (const float*)d_in[7];
  const float* bo = (const float*)d_in[8];
  float* out = (float*)d_out;

  const size_t nX = (size_t)BATCH * SEQ * EMB;
  const size_t nW = (size_t)EMB * EMB;
  const size_t nH = (size_t)BATCH * HEADS * SEQ * HDIM;
  bf16* xb    = (bf16*)d_ws;
  bf16* wqb   = xb  + nX;
  bf16* wkb   = wqb + nW;
  bf16* wvb   = wkb + nW;
  bf16* wob   = wvb + nW;
  bf16* Qb    = wob + nW;
  bf16* Kb    = Qb  + nH;
  bf16* Vtb   = Kb  + nH;
  bf16* attnb = Vtb + nH;

  // 1) fp32 -> bf16
  cvt_f32_bf16_kernel<<<dim3((unsigned)(nX / 4 / 256)), dim3(256), 0, stream>>>(x,  xb,  (int)(nX / 4));
  cvt_f32_bf16_kernel<<<dim3((unsigned)(nW / 4 / 256)), dim3(256), 0, stream>>>(Wq, wqb, (int)(nW / 4));
  cvt_f32_bf16_kernel<<<dim3((unsigned)(nW / 4 / 256)), dim3(256), 0, stream>>>(Wk, wkb, (int)(nW / 4));
  cvt_f32_bf16_kernel<<<dim3((unsigned)(nW / 4 / 256)), dim3(256), 0, stream>>>(Wv, wvb, (int)(nW / 4));
  cvt_f32_bf16_kernel<<<dim3((unsigned)(nW / 4 / 256)), dim3(256), 0, stream>>>(Wo, wob, (int)(nW / 4));

  // 2) Q/K/V projections via TDM-staged WMMA GEMM (Q pre-scaled, V stored transposed)
  dim3 gproj(EMB / 128, (BATCH * SEQ) / 64, 1);
  gemm_tdm_kernel<<<gproj, dim3(128), 0, stream>>>(xb, wqb, bq, Qb,  nullptr, 0);
  gemm_tdm_kernel<<<gproj, dim3(128), 0, stream>>>(xb, wkb, bk, Kb,  nullptr, 1);
  gemm_tdm_kernel<<<gproj, dim3(128), 0, stream>>>(xb, wvb, bv, Vtb, nullptr, 2);

  // 3) banded attention (WMMA + online softmax)
  swa_attn_kernel<<<dim3(SEQ / 16, HEADS, BATCH), dim3(32), 0, stream>>>(Qb, Kb, Vtb, attnb);

  // 4) output projection (fp32 result)
  gemm_tdm_kernel<<<gproj, dim3(128), 0, stream>>>(attnb, wob, bo, nullptr, out, 3);
}
